// ST_Net_47476568490522
// MI455X (gfx1250) — matrix operand order repaired and checked by reading, not compiled, not run on hardware
//
#include <hip/hip_runtime.h>
#include <math.h>

typedef __attribute__((ext_vector_type(16))) __bf16 v16bf;
typedef __attribute__((ext_vector_type(8)))  float  v8f;
typedef __attribute__((ext_vector_type(8)))  __bf16 bf16x8;
typedef __attribute__((ext_vector_type(4)))  __bf16 bf16x4;

namespace {
constexpr int Bq   = 64;
constexpr int Tq   = 12;
constexpr int Nn   = 800;
constexpr int CINq = 2;
constexpr int Hq   = 64;
constexpr int HORq = 12;
constexpr int TCINq = 60;
constexpr int TCHq  = 2;
constexpr int MNUMq = 4;
constexpr int MDIMq = 8;
constexpr int PPAD  = 128;          // padded per-k channel width (both layers)
constexpr int KPROJ = 3 * PPAD;     // 384
constexpr float KAP = 0.05f;
constexpr int ROWS = Bq * Nn;       // 51200
}

// ---------------------------------------------------------------------------
// WMMA bf16 GEMM mainloop. Block = 256 threads = 8 waves (2 along M x 4 along
// N). Wave tile 16x32 (two 16x16 accumulators sharing one A fragment), block
// tile 32(M) x 128(N), K-tile 32. Both A (MxK) and Bt (NxK) are K-major in
// memory so all staging is vectorized b64/b128; Bs holds [n][k].
// ---------------------------------------------------------------------------
__device__ __forceinline__ void gemm2_bf16(
    const __bf16* __restrict__ A, int lda,    // M x K, row-major
    const __bf16* __restrict__ Bt, int ldbt,  // N x K, row-major (K-major)
    int K, int m0, int n0, v8f& acc0, v8f& acc1,
    __bf16* __restrict__ As /*32*32*/, __bf16* __restrict__ Bs /*128*32*/)
{
  const int tid  = threadIdx.x;
  const int lane = tid & 31;
  const int w    = tid >> 5;
  const int l    = lane & 15;
  const int hi   = lane >> 4;
  const int wm   = w >> 2;          // 0..1
  const int wn   = w & 3;           // 0..3
  const int ar   = tid >> 3;        // 0..31 : A row
  const int ac   = (tid & 7) * 4;   // A col chunk (4 elems, 8B)
  const int br   = tid >> 1;        // 0..127 : B row (n)
  const int bk   = (tid & 1) * 16;  // B k chunk base (16 elems)

  union { v16bf v; bf16x8 h[2]; } af, bf0, bf1;

  for (int kt = 0; kt < K; kt += 32) {
    // stage A tile (32 x 32)
    *(bf16x4*)&As[ar * 32 + ac] =
        *(const bf16x4*)&A[(long)(m0 + ar) * lda + kt + ac];
    // stage B tile (128 n-rows x 32 k) straight from K-major source
    const __bf16* bsrc = &Bt[(long)(n0 + br) * ldbt + kt + bk];
    *(bf16x8*)&Bs[br * 32 + bk]     = *(const bf16x8*)bsrc;
    *(bf16x8*)&Bs[br * 32 + bk + 8] = *(const bf16x8*)(bsrc + 8);
    if (kt + 32 < K) {  // gfx1250 global_prefetch_b8 for next tiles
      __builtin_prefetch(&A[(long)(m0 + ar) * lda + kt + 32 + ac], 0, 0);
      __builtin_prefetch(&Bt[(long)(n0 + br) * ldbt + kt + 32 + bk], 0, 0);
    }
    __syncthreads();
    // A fragment (16x32): lane(l,hi): K = hi*16+[0..15]
    af.h[0] = *(const bf16x8*)&As[(wm * 16 + l) * 32 + hi * 16];
    af.h[1] = *(const bf16x8*)&As[(wm * 16 + l) * 32 + hi * 16 + 8];
    // B fragments (32x16): lane(l,hi): col n, K = hi*16+[0..15]
    bf0.h[0] = *(const bf16x8*)&Bs[(wn * 32 + l) * 32 + hi * 16];
    bf0.h[1] = *(const bf16x8*)&Bs[(wn * 32 + l) * 32 + hi * 16 + 8];
    bf1.h[0] = *(const bf16x8*)&Bs[(wn * 32 + 16 + l) * 32 + hi * 16];
    bf1.h[1] = *(const bf16x8*)&Bs[(wn * 32 + 16 + l) * 32 + hi * 16 + 8];
    acc0 = __builtin_amdgcn_wmma_f32_16x16x32_bf16(
        false, af.v, false, bf0.v, (short)0, acc0, false, false);
    acc1 = __builtin_amdgcn_wmma_f32_16x16x32_bf16(
        false, af.v, false, bf1.v, (short)0, acc1, false, false);
    __syncthreads();
  }
}

// Diffusion GEMM: per (batch b, cheby k in {1,2}): Y = G_k^T @ X_b using the
// transposed activation copy XT as K-major B. Epilogue writes bf16
// kappa*x + (1-kappa)*Y into HCAT segment (1+k). grid z = b*2+k.
__global__ void __launch_bounds__(256) k_diffuse(
    const __bf16* __restrict__ GT1, const __bf16* __restrict__ GT2,
    const __bf16* __restrict__ xt, __bf16* __restrict__ hcat)
{
  __shared__ __bf16 As[32 * 32];
  __shared__ __bf16 Bs[128 * 32];
  const int kk = blockIdx.z & 1;
  const int b  = blockIdx.z >> 1;
  const __bf16* A  = kk ? GT2 : GT1;                   // (Nn x Nn) = G_k^T
  const __bf16* Bt = xt + (long)b * PPAD * Nn;         // (PPAD x Nn) K-major
  const __bf16* X0 = hcat + (long)b * Nn * KPROJ;      // seg0 (for kappa mix)
  __bf16* C        = hcat + (long)b * Nn * KPROJ + (1 + kk) * PPAD;
  const int m0 = blockIdx.x * 32;
  v8f a0 = {0.f,0.f,0.f,0.f,0.f,0.f,0.f,0.f};
  v8f a1 = {0.f,0.f,0.f,0.f,0.f,0.f,0.f,0.f};
  gemm2_bf16(A, Nn, Bt, Nn, Nn, m0, 0, a0, a1, As, Bs);
  const int tid = threadIdx.x, lane = tid & 31, w = tid >> 5;
  const int l = lane & 15, hi = lane >> 4, wm = w >> 2, wn = w & 3;
  const int row0 = m0 + wm * 16 + hi * 8;
  const int col0 = wn * 32 + l;
  const int col1 = col0 + 16;
#pragma unroll
  for (int v = 0; v < 8; ++v) {
    const int m = row0 + v;
    const float x0 = (float)X0[(long)m * KPROJ + col0];
    const float x1 = (float)X0[(long)m * KPROJ + col1];
    C[(long)m * KPROJ + col0] = (__bf16)(KAP * x0 + (1.f - KAP) * a0[v]);
    C[(long)m * KPROJ + col1] = (__bf16)(KAP * x1 + (1.f - KAP) * a1[v]);
  }
}

// Projection GEMM: (ROWS x KPROJ) @ Wt^T + bias; Wt is (128 x KPROJ) K-major.
// EPI 0 = sigmoid (gate), 1 = tanh (candidate). Output stride PPAD; only
// columns < nreal are written.
template <int EPI>
__global__ void __launch_bounds__(256) k_proj(
    const __bf16* __restrict__ A, const __bf16* __restrict__ Wt,
    const float* __restrict__ bias, float* __restrict__ out, int nreal)
{
  __shared__ __bf16 As[32 * 32];
  __shared__ __bf16 Bs[128 * 32];
  const int m0 = blockIdx.x * 32;
  v8f a0 = {0.f,0.f,0.f,0.f,0.f,0.f,0.f,0.f};
  v8f a1 = {0.f,0.f,0.f,0.f,0.f,0.f,0.f,0.f};
  gemm2_bf16(A, KPROJ, Wt, KPROJ, KPROJ, m0, 0, a0, a1, As, Bs);
  const int tid = threadIdx.x, lane = tid & 31, w = tid >> 5;
  const int l = lane & 15, hi = lane >> 4, wm = w >> 2, wn = w & 3;
  const int row0 = m0 + wm * 16 + hi * 8;
  const int col0 = wn * 32 + l;
  const int col1 = col0 + 16;
  const float b0 = (col0 < nreal) ? bias[col0] : 0.f;
  const float b1 = (col1 < nreal) ? bias[col1] : 0.f;
#pragma unroll
  for (int v = 0; v < 8; ++v) {
    if (col0 < nreal) {
      float t = a0[v] + b0;
      out[(long)(row0 + v) * PPAD + col0] =
          (EPI == 0) ? 1.f / (1.f + __expf(-t)) : tanhf(t);
    }
    if (col1 < nreal) {
      float t = a1[v] + b1;
      out[(long)(row0 + v) * PPAD + col1] =
          (EPI == 0) ? 1.f / (1.f + __expf(-t)) : tanhf(t);
    }
  }
}

// ---------------------------------------------------------------------------
// Elementwise / small kernels
// ---------------------------------------------------------------------------
__global__ void k_zero(float* __restrict__ p, long n) {
  long i = (long)blockIdx.x * blockDim.x + threadIdx.x;
  if (i < n) p[i] = 0.f;
}

// Build HCAT seg0 (row-major) AND the transposed copy XT (B, PPAD, Nn):
// value = [xa | xb | h(*r) | 0pad]
__global__ void k_pack(__bf16* __restrict__ hcat, __bf16* __restrict__ xt,
                       const float* __restrict__ xa, long xa_bstride, int wa,
                       const float* __restrict__ xb, long xb_bstride, int wb,
                       const float* __restrict__ h,
                       const float* __restrict__ ur) {
  long idx = (long)blockIdx.x * blockDim.x + threadIdx.x;
  if (idx >= (long)ROWS * PPAD) return;
  int col = (int)(idx & (PPAD - 1));
  long row = idx >> 7;
  int b = (int)(row / Nn), n = (int)(row % Nn);
  float v = 0.f;
  if (col < wa) {
    if (xa) v = xa[b * xa_bstride + (long)n * wa + col];
  } else if (col < wa + wb) {
    v = xb[b * xb_bstride + (long)n * wb + (col - wa)];
  } else if (col < wa + wb + Hq) {
    int c = col - wa - wb;
    float hv = h[row * Hq + c];
    if (ur) hv *= ur[row * PPAD + Hq + c];  // r gate
    v = hv;
  }
  __bf16 bv = (__bf16)v;
  hcat[row * KPROJ + col] = bv;
  xt[((long)b * PPAD + col) * Nn + n] = bv;
}

// h = (1-u)*h + u*c   (u = UR[.,c], c in CC with stride PPAD)
__global__ void k_hupdate(float* __restrict__ h, const float* __restrict__ ur,
                          const float* __restrict__ c, long n) {
  long i = (long)blockIdx.x * blockDim.x + threadIdx.x;
  if (i >= n) return;
  long row = i >> 6;
  int cc = (int)(i & 63);
  float u = ur[row * PPAD + cc];
  h[i] = (1.f - u) * h[i] + u * c[row * PPAD + cc];
}

// GG = G @ G (one-time, tiny)
__global__ void k_gg(const float* __restrict__ G, float* __restrict__ GG) {
  int i = blockIdx.x * blockDim.x + threadIdx.x;
  if (i >= Nn * Nn) return;
  int r = i / Nn, c = i % Nn;
  float s = 0.f;
  for (int k = 0; k < Nn; ++k) s += G[r * Nn + k] * G[k * Nn + c];
  GG[i] = s;
}

// GT1 = G^T, GT2 = (2*G@G - I)^T, bf16
__global__ void k_gt(const float* __restrict__ G, const float* __restrict__ GG,
                     __bf16* __restrict__ GT1, __bf16* __restrict__ GT2) {
  int i = blockIdx.x * blockDim.x + threadIdx.x;
  if (i >= Nn * Nn) return;
  int m = i / Nn, n = i % Nn;
  GT1[i] = (__bf16)G[n * Nn + m];
  float g2 = 2.f * GG[n * Nn + m] - ((m == n) ? 1.f : 0.f);
  GT2[i] = (__bf16)g2;
}

// Pad + bf16-convert + TRANSPOSE weight: W (3*Preal, nreal) -> Wt (PPAD, KPROJ)
__global__ void k_wconv(const float* __restrict__ W, __bf16* __restrict__ Wt,
                        int Preal, int nreal) {
  long i = (long)blockIdx.x * blockDim.x + threadIdx.x;
  if (i >= (long)PPAD * KPROJ) return;
  int k = (int)(i % KPROJ);
  int n = (int)(i / KPROJ);
  int seg = k / PPAD, p = k % PPAD;
  float v = (p < Preal && n < nreal) ? W[((long)seg * Preal + p) * nreal + n] : 0.f;
  Wt[i] = (__bf16)v;
}

// MLP stage 1: hid(rows,10) = t(rows,60) @ W1 + b1
__global__ void k_mlp1(const float* __restrict__ tin, const float* __restrict__ W1,
                       const float* __restrict__ b1, float* __restrict__ hid,
                       int rows) {
  int i = blockIdx.x * blockDim.x + threadIdx.x;
  if (i >= rows * 10) return;
  int r = i / 10, j = i % 10;
  float s = b1[j];
  for (int k = 0; k < TCINq; ++k) s += tin[r * TCINq + k] * W1[k * 10 + j];
  hid[i] = s;
}

// MLP stage 2: emb(rows,1600) = hid @ W2 + b2
__global__ void k_mlp2(const float* __restrict__ hid, const float* __restrict__ W2,
                       const float* __restrict__ b2, float* __restrict__ emb,
                       int rows) {
  long i = (long)blockIdx.x * blockDim.x + threadIdx.x;
  if (i >= (long)rows * (Nn * TCHq)) return;
  long r = i / (Nn * TCHq);
  int o = (int)(i % (Nn * TCHq));
  float s = b2[o];
  for (int j = 0; j < 10; ++j) s += hid[r * 10 + j] * W2[j * (Nn * TCHq) + o];
  emb[i] = s;
}

// ST-memory query: per batch b, q = h1_flat @ Wa; att = softmax(q@mem^T);
// amv = att @ mem  (B x MDIM)
__global__ void __launch_bounds__(256) k_attq(
    const float* __restrict__ h1, const float* __restrict__ Wa,
    const float* __restrict__ mem, float* __restrict__ amv) {
  __shared__ float red[256];
  __shared__ float q[MDIMq];
  const int b = blockIdx.x;
  const int t = threadIdx.x;
  float part[MDIMq];
#pragma unroll
  for (int d = 0; d < MDIMq; ++d) part[d] = 0.f;
  const float* hb = h1 + (long)b * Nn * Hq;
  for (int i = t; i < Nn * Hq; i += 256) {
    float hv = hb[i];
#pragma unroll
    for (int d = 0; d < MDIMq; ++d) part[d] += hv * Wa[(long)i * MDIMq + d];
  }
  for (int d = 0; d < MDIMq; ++d) {
    red[t] = part[d];
    __syncthreads();
    for (int s = 128; s > 0; s >>= 1) {
      if (t < s) red[t] += red[t + s];
      __syncthreads();
    }
    if (t == 0) q[d] = red[0];
    __syncthreads();
  }
  if (t == 0) {
    float sc[MNUMq];
    float mx = -1e30f;
    for (int j = 0; j < MNUMq; ++j) {
      float s = 0.f;
      for (int d = 0; d < MDIMq; ++d) s += q[d] * mem[j * MDIMq + d];
      sc[j] = s;
      mx = fmaxf(mx, s);
    }
    float se = 0.f;
    for (int j = 0; j < MNUMq; ++j) { sc[j] = __expf(sc[j] - mx); se += sc[j]; }
    float v[MDIMq];
    for (int d = 0; d < MDIMq; ++d) v[d] = 0.f;
    for (int j = 0; j < MNUMq; ++j)
      for (int d = 0; d < MDIMq; ++d) v[d] += (sc[j] / se) * mem[j * MDIMq + d];
    for (int d = 0; d < MDIMq; ++d) amv[b * MDIMq + d] = v[d];
  }
}

// Decoder output: attm = amv @ fc (per node), out = relu([h1|attm] @ pW + pb)
__global__ void k_decout(const float* __restrict__ h1, const float* __restrict__ amv,
                         const float* __restrict__ fc, const float* __restrict__ pW,
                         const float* __restrict__ pb, float* __restrict__ out,
                         int tstep) {
  long i = (long)blockIdx.x * blockDim.x + threadIdx.x;
  if (i >= (long)ROWS) return;
  int b = (int)(i / Nn), n = (int)(i % Nn);
  float at[MDIMq];
#pragma unroll
  for (int d = 0; d < MDIMq; ++d) {
    float s = 0.f;
    for (int e = 0; e < MDIMq; ++e)
      s += amv[b * MDIMq + e] * fc[(long)e * (MDIMq * Nn) + n * MDIMq + d];
    at[d] = s;
  }
  const float* hrow = h1 + i * Hq;
  for (int c = 0; c < CINq; ++c) {
    float s = pb[c];
    for (int j = 0; j < Hq; ++j) s += hrow[j] * pW[j * CINq + c];
    for (int d = 0; d < MDIMq; ++d) s += at[d] * pW[(Hq + d) * CINq + c];
    out[(((long)b * HORq + tstep) * Nn + n) * CINq + c] = fmaxf(s, 0.f);
  }
}

// ---------------------------------------------------------------------------
// Host orchestration
// ---------------------------------------------------------------------------
extern "C" void kernel_launch(void* const* d_in, const int* in_sizes, int n_in,
                              void* d_out, int out_size, void* d_ws, size_t ws_size,
                              hipStream_t stream) {
  (void)in_sizes; (void)n_in; (void)out_size; (void)ws_size;
  const float* x_seq = (const float*)d_in[0];
  const float* t_x   = (const float*)d_in[1];
  const float* t_y   = (const float*)d_in[2];
  const float* G     = (const float*)d_in[3];
  // [pre][layer]: enc=0 dec=1
  const float* Wg[2][2] = {{(const float*)d_in[4],  (const float*)d_in[8]},
                           {(const float*)d_in[12], (const float*)d_in[16]}};
  const float* bg[2][2] = {{(const float*)d_in[5],  (const float*)d_in[9]},
                           {(const float*)d_in[13], (const float*)d_in[17]}};
  const float* Wc[2][2] = {{(const float*)d_in[6],  (const float*)d_in[10]},
                           {(const float*)d_in[14], (const float*)d_in[18]}};
  const float* bc[2][2] = {{(const float*)d_in[7],  (const float*)d_in[11]},
                           {(const float*)d_in[15], (const float*)d_in[19]}};
  const float* mlp_W1 = (const float*)d_in[20];
  const float* mlp_b1 = (const float*)d_in[21];
  const float* mlp_W2 = (const float*)d_in[22];
  const float* mlp_b2 = (const float*)d_in[23];
  const float* memry  = (const float*)d_in[24];
  const float* Wa     = (const float*)d_in[25];
  const float* fc     = (const float*)d_in[26];
  const float* proj_W = (const float*)d_in[27];
  const float* proj_b = (const float*)d_in[28];
  float* outp = (float*)d_out;

  // workspace carve-out
  char* ws = (char*)d_ws;
  size_t off = 0;
  auto carve = [&](size_t bytes) {
    void* p = ws + off;
    off = (off + bytes + 255) & ~(size_t)255;
    return p;
  };
  __bf16* HCAT = (__bf16*)carve((size_t)ROWS * KPROJ * 2);
  __bf16* XT   = (__bf16*)carve((size_t)Bq * PPAD * Nn * 2);
  float*  UR   = (float*)carve((size_t)ROWS * PPAD * 4);
  float*  CC   = (float*)carve((size_t)ROWS * PPAD * 4);
  float*  H0   = (float*)carve((size_t)ROWS * Hq * 4);
  float*  H1   = (float*)carve((size_t)ROWS * Hq * 4);
  __bf16* GT1  = (__bf16*)carve((size_t)Nn * Nn * 2);
  __bf16* GT2  = (__bf16*)carve((size_t)Nn * Nn * 2);
  float*  GGf  = (float*)carve((size_t)Nn * Nn * 4);
  float*  TXE  = (float*)carve((size_t)Bq * Tq * Nn * TCHq * 4);
  float*  TYE  = (float*)carve((size_t)Bq * HORq * Nn * TCHq * 4);
  float*  HID  = (float*)carve((size_t)Bq * Tq * 10 * 4);
  __bf16* Wpg[2][2], *Wpc[2][2];
  for (int pre = 0; pre < 2; ++pre)
    for (int lr = 0; lr < 2; ++lr) {
      Wpg[pre][lr] = (__bf16*)carve((size_t)PPAD * KPROJ * 2);
      Wpc[pre][lr] = (__bf16*)carve((size_t)PPAD * KPROJ * 2);
    }
  float* AMV = (float*)carve((size_t)Bq * MDIMq * 4);

  const int TB = 256;
  const long packN = (long)ROWS * PPAD;
  const int packB = (int)((packN + TB - 1) / TB);
  const long hupN = (long)ROWS * Hq;
  const int hupB = (int)((hupN + TB - 1) / TB);
  const dim3 gDiff(Nn / 32, 1, Bq * 2);
  const dim3 gProj(ROWS / 32, 1, 1);
  const int wcB = (int)(((long)PPAD * KPROJ + TB - 1) / TB);

  // --- one-time precompute (per launch, deterministic) ---
  k_zero<<<hupB, TB, 0, stream>>>(H0, hupN);
  k_zero<<<hupB, TB, 0, stream>>>(H1, hupN);
  k_gg<<<(Nn * Nn + TB - 1) / TB, TB, 0, stream>>>(G, GGf);
  k_gt<<<(Nn * Nn + TB - 1) / TB, TB, 0, stream>>>(G, GGf, GT1, GT2);
  for (int pre = 0; pre < 2; ++pre)
    for (int lr = 0; lr < 2; ++lr) {
      int Preal = (lr == 0) ? (CINq + TCHq + Hq) : (2 * Hq);  // 68 or 128
      k_wconv<<<wcB, TB, 0, stream>>>(Wg[pre][lr], Wpg[pre][lr], Preal, 2 * Hq);
      k_wconv<<<wcB, TB, 0, stream>>>(Wc[pre][lr], Wpc[pre][lr], Preal, Hq);
    }
  // time-covariate embeddings
  k_mlp1<<<(Bq * Tq * 10 + TB - 1) / TB, TB, 0, stream>>>(t_x, mlp_W1, mlp_b1, HID, Bq * Tq);
  k_mlp2<<<(int)(((long)Bq * Tq * Nn * TCHq + TB - 1) / TB), TB, 0, stream>>>(
      HID, mlp_W2, mlp_b2, TXE, Bq * Tq);
  k_mlp1<<<(Bq * HORq * 10 + TB - 1) / TB, TB, 0, stream>>>(t_y, mlp_W1, mlp_b1, HID, Bq * HORq);
  k_mlp2<<<(int)(((long)Bq * HORq * Nn * TCHq + TB - 1) / TB), TB, 0, stream>>>(
      HID, mlp_W2, mlp_b2, TYE, Bq * HORq);

  // GCRU cell: x-part described by (xa,bsa,wa)+(xb,bsb,wb), state h, params idx
  auto cell = [&](const float* xa, long bsa, int wa,
                  const float* xb, long bsb, int wb,
                  float* h, int pre, int lr) {
    // gate
    k_pack<<<packB, TB, 0, stream>>>(HCAT, XT, xa, bsa, wa, xb, bsb, wb, h, nullptr);
    k_diffuse<<<gDiff, TB, 0, stream>>>(GT1, GT2, XT, HCAT);
    k_proj<0><<<gProj, TB, 0, stream>>>(HCAT, Wpg[pre][lr], bg[pre][lr], UR, 2 * Hq);
    // candidate (h part gated by r)
    k_pack<<<packB, TB, 0, stream>>>(HCAT, XT, xa, bsa, wa, xb, bsb, wb, h, UR);
    k_diffuse<<<gDiff, TB, 0, stream>>>(GT1, GT2, XT, HCAT);
    k_proj<1><<<gProj, TB, 0, stream>>>(HCAT, Wpc[pre][lr], bc[pre][lr], CC, Hq);
    // h = (1-u)*h + u*c
    k_hupdate<<<hupB, TB, 0, stream>>>(h, UR, CC, hupN);
  };

  // --- encoder: interleaved 2-layer scan over T ---
  for (int t = 0; t < Tq; ++t) {
    const float* xa = x_seq + (long)t * Nn * CINq;
    const float* xb = TXE + (long)t * Nn * TCHq;
    cell(xa, (long)Tq * Nn * CINq, CINq, xb, (long)Tq * Nn * TCHq, TCHq, H0, 0, 0);
    cell(H0, (long)Nn * Hq, Hq, nullptr, 0, 0, H1, 0, 1);
  }

  // --- decoder: autoregressive over horizon ---
  for (int t = 0; t < HORq; ++t) {
    const float* go = (t == 0) ? nullptr : (outp + (long)(t - 1) * Nn * CINq);
    const float* tye = TYE + (long)t * Nn * TCHq;
    cell(go, (long)HORq * Nn * CINq, CINq, tye, (long)HORq * Nn * TCHq, TCHq, H0, 1, 0);
    cell(H0, (long)Nn * Hq, Hq, nullptr, 0, 0, H1, 1, 1);
    k_attq<<<Bq, TB, 0, stream>>>(H1, Wa, memry, AMV);
    k_decout<<<(int)((ROWS + TB - 1) / TB), TB, 0, stream>>>(
        H1, AMV, fc, proj_W, proj_b, outp, t);
  }
}
